// part_seg2_15264313770019
// MI455X (gfx1250) — matrix operand order repaired and checked by reading, not compiled, not run on hardware
//
#include <hip/hip_runtime.h>

// ---------------------------------------------------------------------------
// GCN pipeline for MI455X (gfx1250, wave32):
//   deg/dinv -> scalar MLP (3->32->64->128) -> 3x [bf16 WMMA GEMM (LDS-staged
//   B panel, 4 N-tiles per wave) -> edge scatter (float atomics) ->
//   finalize(+self-loop,+bias,ReLU,bf16)] -> softmax
// ---------------------------------------------------------------------------

typedef __attribute__((ext_vector_type(16))) __bf16 v16bf;
typedef __attribute__((ext_vector_type(8)))  float  v8f;

__device__ __forceinline__ unsigned short f2bf(float f) {
  unsigned u = __builtin_bit_cast(unsigned, f);
  u += 0x7FFFu + ((u >> 16) & 1u);   // round-to-nearest-even
  return (unsigned short)(u >> 16);
}

// -------------------------- small utility kernels --------------------------

__global__ void k_zero_f32(float* __restrict__ p, long long n) {
  long long i = (long long)blockIdx.x * blockDim.x + threadIdx.x;
  if (i < n) p[i] = 0.0f;
}

__global__ void k_deg(const long long* __restrict__ ei, float* __restrict__ deg, int E) {
  int e = blockIdx.x * blockDim.x + threadIdx.x;
  if (e < E) {
    int d = (int)ei[E + e];       // dst row of edge_index
    atomicAdd(deg + d, 1.0f);
  }
}

__global__ void k_dinv(float* __restrict__ deg, int n) {
  int i = blockIdx.x * blockDim.x + threadIdx.x;
  if (i < n) deg[i] = rsqrtf(deg[i] + 1.0f);   // +1 self-loop
}

// W [K,N] row-major (f32) -> Wt [Npad,K] bf16 (pre-transposed, zero-padded so
// GEMM B loads need no guards). Thread per (n,k) of the padded matrix.
__global__ void k_cast_transpose_pad(const float* __restrict__ W, unsigned short* __restrict__ Wt,
                                     int K, int N, int Npad) {
  long long idx = (long long)blockIdx.x * blockDim.x + threadIdx.x;
  if (idx >= (long long)Npad * K) return;
  int n = (int)(idx / K);
  int k = (int)(idx % K);
  float v = (n < N) ? W[(size_t)k * N + n] : 0.0f;
  Wt[(size_t)n * K + k] = f2bf(v);
}

// ------------------------------ pointwise MLP -------------------------------
// 3 -> 32 -> 64 -> 128 with ReLU; weights are wave-uniform (scalar cache).
__global__ void k_mlp(const float* __restrict__ x,
                      const float* __restrict__ w1, const float* __restrict__ b1,
                      const float* __restrict__ w2, const float* __restrict__ b2,
                      const float* __restrict__ w3, const float* __restrict__ b3,
                      unsigned short* __restrict__ hbf, int Nn) {
  int i = blockIdx.x * blockDim.x + threadIdx.x;
  if (i >= Nn) return;
  float x0 = x[3 * i + 0], x1 = x[3 * i + 1], x2 = x[3 * i + 2];
  float h1[32];
#pragma unroll
  for (int j = 0; j < 32; ++j) {
    float a = x0 * w1[j] + x1 * w1[32 + j] + x2 * w1[64 + j] + b1[j];
    h1[j] = fmaxf(a, 0.0f);
  }
  float h2[64];
  for (int j = 0; j < 64; ++j) {
    float a = b2[j];
#pragma unroll 8
    for (int k = 0; k < 32; ++k) a += h1[k] * w2[k * 64 + j];
    h2[j] = fmaxf(a, 0.0f);
  }
  unsigned short* out = hbf + (size_t)i * 128;
  for (int j = 0; j < 128; ++j) {
    float a = b3[j];
#pragma unroll 8
    for (int k = 0; k < 64; ++k) a += h2[k] * w3[k * 128 + j];
    out[j] = f2bf(fmaxf(a, 0.0f));
  }
}

// ------------------------------- WMMA GEMM ----------------------------------
// C[M,N] (f32) = A[M,K] (bf16 row-major) x Wt[Npad,K] (bf16, pre-transposed).
// Block = 8 waves. Block covers 8 M-tiles x one 64-wide N panel.
// The B panel (64 x K bf16) is staged in LDS with +8-element row padding
// (row stride 260 dwords -> ds_load_b128 lane banks 4L..4L+3, conflict-free).
// Each wave: one A fragment reused across 4 WMMAs (4 N-tiles) per K-step.
__global__ void k_gemm_bf16_wmma(const unsigned short* __restrict__ A,
                                 const unsigned short* __restrict__ Wt,
                                 float* __restrict__ C, int M, int N, int K) {
  extern __shared__ unsigned short sB[];          // [64][K+8]
  const int pitch = K + 8;
  const int waveid = threadIdx.x >> 5;
  const int lane   = threadIdx.x & 31;
  const int mtiles = M >> 4;
  const int tn_base = blockIdx.y * 64;

  // ---- cooperative stage of B panel: wave w loads rows 8w..8w+7 ----
  {
    const int kc8 = K >> 3;                        // float4 chunks per row
    for (int r = 0; r < 8; ++r) {
      int row = waveid * 8 + r;
      const float4* src = (const float4*)(Wt + (size_t)(tn_base + row) * K);
      float4* dst = (float4*)(sB + (size_t)row * pitch);
      for (int kc = lane; kc < kc8; kc += 32) dst[kc] = src[kc];
    }
  }
  __syncthreads();

  const int tm = blockIdx.x * 8 + waveid;
  if (tm < mtiles) {                               // wave-uniform guard
    const int half = lane >> 4;                    // 0: lanes 0-15, 1: lanes 16-31
    const int row  = tm * 16 + (lane & 15);        // A row (M)
    const int offA = half ? 16 : 0;                // A frag: K 0..15 vs 16..31
    const int offB = half ? 16 : 0;                // B frag: K 0..15 vs 16..31
    const unsigned short* arow = A + (size_t)row * K;
    const unsigned short* brow = sB + (size_t)(lane & 15) * pitch + offB;

    v8f acc0 = {}, acc1 = {}, acc2 = {}, acc3 = {};
    for (int kb = 0; kb < K; kb += 32) {
      union { v16bf v; float4 f[2]; } a, b0, b1, b2, b3;
      a.f[0] = *(const float4*)(arow + kb + offA);
      a.f[1] = *(const float4*)(arow + kb + offA + 8);
      __builtin_prefetch(arow + kb + 32, 0, 1);    // speculative next-A chunk
      const unsigned short* bk = brow + kb;
      b0.f[0] = *(const float4*)(bk);
      b0.f[1] = *(const float4*)(bk + 8);
      b1.f[0] = *(const float4*)(bk + 16 * pitch);
      b1.f[1] = *(const float4*)(bk + 16 * pitch + 8);
      b2.f[0] = *(const float4*)(bk + 32 * pitch);
      b2.f[1] = *(const float4*)(bk + 32 * pitch + 8);
      b3.f[0] = *(const float4*)(bk + 48 * pitch);
      b3.f[1] = *(const float4*)(bk + 48 * pitch + 8);
      acc0 = __builtin_amdgcn_wmma_f32_16x16x32_bf16(false, a.v, false, b0.v, (short)0, acc0, false, false);
      acc1 = __builtin_amdgcn_wmma_f32_16x16x32_bf16(false, a.v, false, b1.v, (short)0, acc1, false, false);
      acc2 = __builtin_amdgcn_wmma_f32_16x16x32_bf16(false, a.v, false, b2.v, (short)0, acc2, false, false);
      acc3 = __builtin_amdgcn_wmma_f32_16x16x32_bf16(false, a.v, false, b3.v, (short)0, acc3, false, false);
    }

    // C/D layout: lanes 0-15 -> M=tm*16+r, lanes 16-31 -> M=tm*16+8+r; N = col
    const int mbase = tm * 16 + half * 8;
    v8f accs[4] = {acc0, acc1, acc2, acc3};
#pragma unroll
    for (int t = 0; t < 4; ++t) {
      int n = tn_base + t * 16 + (lane & 15);
      if (n < N) {
#pragma unroll
        for (int r = 0; r < 8; ++r)
          C[(size_t)(mbase + r) * N + n] = accs[t][r];
      }
    }
  }
}

// ---------------------------- edge gather/scatter ---------------------------
// agg[dst] += hw[src] * dinv[src]*dinv[dst]; blockIdx.y selects feature chunk.
template <int VEC>
__global__ void k_scatter(const float* __restrict__ hw, const long long* __restrict__ ei,
                          const float* __restrict__ dinv, float* __restrict__ agg,
                          int E, int F) {
  int e = blockIdx.x * blockDim.x + threadIdx.x;
  if (e >= E) return;
  int c = blockIdx.y * VEC;
  int s = (int)ei[e];
  int d = (int)ei[E + e];
  float norm = dinv[s] * dinv[d];
  const float* p = hw + (size_t)s * F + c;
  float* q = agg + (size_t)d * F + c;
#pragma unroll
  for (int i = 0; i < VEC; ++i) atomicAdd(q + i, p[i] * norm);
}

// ----------------------------- layer finalize -------------------------------
// v = agg + hw*dinv^2 + bias ; optional ReLU; optional f32 and/or bf16 outputs.
__global__ void k_finalize(const float* __restrict__ agg, const float* __restrict__ hw,
                           const float* __restrict__ dinv, const float* __restrict__ bias,
                           float* __restrict__ out_f32, unsigned short* __restrict__ out_bf,
                           int Nn, int F, int do_relu) {
  long long idx = (long long)blockIdx.x * blockDim.x + threadIdx.x;
  if (idx >= (long long)Nn * F) return;
  int i = (int)(idx / F);
  int j = (int)(idx % F);
  float di = dinv[i];
  float v = agg[idx] + hw[idx] * di * di + bias[j];
  if (do_relu) v = fmaxf(v, 0.0f);
  if (out_f32) out_f32[idx] = v;
  if (out_bf)  out_bf[idx]  = f2bf(v);
}

// --------------------------------- softmax ----------------------------------
__global__ void k_softmax(const float* __restrict__ in, float* __restrict__ out,
                          int Nn, int F) {
  int i = blockIdx.x * blockDim.x + threadIdx.x;
  if (i >= Nn) return;
  const float* r = in + (size_t)i * F;
  float m = -3.4e38f;
  for (int j = 0; j < F; ++j) m = fmaxf(m, r[j]);
  float s = 0.0f;
  float* o = out + (size_t)i * F;
  for (int j = 0; j < F; ++j) { float e = __expf(r[j] - m); o[j] = e; s += e; }
  float inv = __frcp_rn(s);
  for (int j = 0; j < F; ++j) o[j] *= inv;
}

// ------------------------------- host driver --------------------------------

static inline size_t align256(size_t x) { return (x + 255) & ~(size_t)255; }
static inline int pad64(int n) { return (n + 63) & ~63; }

static void run_gcn_layer(const unsigned short* hbf_in, const unsigned short* wt,
                          const float* bias, const long long* ei, const float* dinv,
                          float* hw, float* agg,
                          float* out_f32, unsigned short* out_bf,
                          int Nn, int E, int Fin, int Fout, int do_relu,
                          hipStream_t stream) {
  // GEMM: hw = h @ W   (8 M-tiles x 64-wide N panel per block)
  {
    int mtiles = Nn >> 4;
    dim3 grid((mtiles + 7) / 8, pad64(Fout) / 64, 1);
    size_t smem = (size_t)64 * (Fin + 8) * sizeof(unsigned short);
    k_gemm_bf16_wmma<<<grid, 256, smem, stream>>>(hbf_in, wt, hw, Nn, Fout, Fin);
  }
  // zero aggregate
  {
    long long n = (long long)Nn * Fout;
    k_zero_f32<<<(unsigned)((n + 255) / 256), 256, 0, stream>>>(agg, n);
  }
  // edge scatter
  {
    dim3 grid((E + 255) / 256, 1, 1);
    if ((Fout & 3) == 0) {
      grid.y = Fout / 4;
      k_scatter<4><<<grid, 256, 0, stream>>>(hw, ei, dinv, agg, E, Fout);
    } else {
      grid.y = Fout / 2;
      k_scatter<2><<<grid, 256, 0, stream>>>(hw, ei, dinv, agg, E, Fout);
    }
  }
  // finalize
  {
    long long n = (long long)Nn * Fout;
    k_finalize<<<(unsigned)((n + 255) / 256), 256, 0, stream>>>(
        agg, hw, dinv, bias, out_f32, out_bf, Nn, Fout, do_relu);
  }
}

extern "C" void kernel_launch(void* const* d_in, const int* in_sizes, int n_in,
                              void* d_out, int out_size, void* d_ws, size_t ws_size,
                              hipStream_t stream) {
  const float*      x   = (const float*)d_in[0];
  const long long*  ei  = (const long long*)d_in[1];   // int64 edge_index [2,E]
  const float* w1  = (const float*)d_in[2];
  const float* b1  = (const float*)d_in[3];
  const float* w2  = (const float*)d_in[4];
  const float* b2  = (const float*)d_in[5];
  const float* w3  = (const float*)d_in[6];
  const float* b3  = (const float*)d_in[7];
  const float* g1w = (const float*)d_in[8];
  const float* g1b = (const float*)d_in[9];
  const float* g2w = (const float*)d_in[10];
  const float* g2b = (const float*)d_in[11];
  const float* g3w = (const float*)d_in[12];
  const float* g3b = (const float*)d_in[13];

  const int Nn = in_sizes[0] / 3;     // 50000 (multiple of 16)
  const int E  = in_sizes[1] / 2;     // 800000

  // ---- workspace layout ----
  char* w = (char*)d_ws;
  float* dinv = (float*)w;                  w += align256((size_t)Nn * 4);
  unsigned short* hbf = (unsigned short*)w; w += align256((size_t)Nn * 512 * 2);
  float* hw   = (float*)w;                  w += align256((size_t)Nn * 512 * 4);
  float* agg  = (float*)w;                  w += align256((size_t)Nn * 512 * 4);
  unsigned short* wt1 = (unsigned short*)w; w += align256((size_t)pad64(256) * 128 * 2);
  unsigned short* wt2 = (unsigned short*)w; w += align256((size_t)pad64(512) * 256 * 2);
  unsigned short* wt3 = (unsigned short*)w; w += align256((size_t)pad64(50)  * 512 * 2);
  (void)ws_size;

  // ---- degree / dinv ----
  k_zero_f32<<<(Nn + 255) / 256, 256, 0, stream>>>(dinv, Nn);
  k_deg<<<(E + 255) / 256, 256, 0, stream>>>(ei, dinv, E);
  k_dinv<<<(Nn + 255) / 256, 256, 0, stream>>>(dinv, Nn);

  // ---- weight conversion (f32 [K,N] -> bf16 [Npad,K], zero-padded) ----
  {
    long long n1 = (long long)pad64(256) * 128;
    long long n2 = (long long)pad64(512) * 256;
    long long n3 = (long long)pad64(50)  * 512;
    k_cast_transpose_pad<<<(unsigned)((n1 + 255) / 256), 256, 0, stream>>>(g1w, wt1, 128, 256, pad64(256));
    k_cast_transpose_pad<<<(unsigned)((n2 + 255) / 256), 256, 0, stream>>>(g2w, wt2, 256, 512, pad64(512));
    k_cast_transpose_pad<<<(unsigned)((n3 + 255) / 256), 256, 0, stream>>>(g3w, wt3, 512, 50,  pad64(50));
  }

  // ---- pointwise MLP 3->32->64->128 (writes bf16 features) ----
  k_mlp<<<(Nn + 255) / 256, 256, 0, stream>>>(x, w1, b1, w2, b2, w3, b3, hbf, Nn);

  // ---- GCN layers ----
  run_gcn_layer(hbf, wt1, g1b, ei, dinv, hw, agg, nullptr, hbf, Nn, E, 128, 256, 1, stream);
  run_gcn_layer(hbf, wt2, g2b, ei, dinv, hw, agg, nullptr, hbf, Nn, E, 256, 512, 1, stream);
  run_gcn_layer(hbf, wt3, g3b, ei, dinv, hw, agg, agg, nullptr, Nn, E, 512, 50, 0, stream);

  // ---- softmax over 50 classes -> d_out ----
  k_softmax<<<(Nn + 255) / 256, 256, 0, stream>>>(agg, (float*)d_out, Nn, 50);
}
